// MLTSSD_encoding_13855564497376
// MI455X (gfx1250) — compile-verified, matrix operand order checked by reading, not run on hardware
//
#include <hip/hip_runtime.h>
#include <cstdint>

// ---------------------------------------------------------------------------
// MLT-SSD encoding pipeline for MI455X (gfx1250, wave32, WMMA bf16)
// ---------------------------------------------------------------------------
#define B_      2
#define NP_     200000
#define N_      (B_ * NP_)
#define H_      512
#define W_      512
#define C_      64
#define S_      (B_ * H_ * W_)
#define NPOINT_ 16384
#define NCLS_   20

typedef __attribute__((ext_vector_type(16))) __bf16        v16bf;
typedef __attribute__((ext_vector_type(8)))  float         v8f;
typedef __attribute__((ext_vector_type(8)))  unsigned int  v8u;

// K-index inside a 16x32 A-tile / 32x16 B-tile for VGPR pair j (0..7), half = lane>>4
__device__ __forceinline__ int frag_kk(int j, int half) {
  return (j < 4 ? 2 * j : 16 + 2 * (j - 4)) + half * 8;
}

__device__ __forceinline__ unsigned pack_bf16x2(float lo, float hi) {
  unsigned short l = __builtin_bit_cast(unsigned short, (__bf16)lo);
  unsigned short h = __builtin_bit_cast(unsigned short, (__bf16)hi);
  return (unsigned)l | ((unsigned)h << 16);
}

// ---------------------------------------------------------------------------
// Weight packer: src is KxN f32 row-major -> dst [K/32][NT][32 lanes][16] bf16
// (B-fragment layout, N zero-padded to NT*16)
// ---------------------------------------------------------------------------
__global__ void pack_b_kernel(const float* __restrict__ src, __bf16* __restrict__ dst,
                              int K, int N, int NT) {
  int total = (K / 32) * NT * 512;
  for (int i = blockIdx.x * 256 + threadIdx.x; i < total; i += gridDim.x * 256) {
    int i16  = i & 15;
    int lane = (i >> 4) & 31;
    int rest = i >> 9;
    int nt   = rest % NT;
    int kc   = rest / NT;
    int half = lane >> 4, col = lane & 15;
    int j = i16 >> 1, lo = i16 & 1;
    int kk = frag_kk(j, half) + lo;
    int k  = kc * 32 + kk;
    int n  = nt * 16 + col;
    float v = (n < N) ? src[(size_t)k * N + n] : 0.0f;
    dst[i] = (__bf16)v;
  }
}

// ---------------------------------------------------------------------------
// Per-point MLPs (bone/sem/det): 4 -> 32 (VALU f32) -> 64 (WMMA bf16)
// One wave handles 32 points (two 16-row A tiles).
// NOTE: the per-MLP body is a forceinline function called with three distinct
// pointer sets so address-space inference keeps global_* (not flat_*) memops.
// ---------------------------------------------------------------------------
__device__ __forceinline__ void mlp_stage(
    const float* __restrict__ w1, const __bf16* __restrict__ p2, float* __restrict__ out,
    float fin0, float fin1, float fin2, float fin3,
    int wid, int lane, int base, __bf16 (*hbuf)[32][32]) {
  // layer1: 4 -> 32, one point per lane, packed u32 LDS stores
  unsigned* hrow = (unsigned*)&hbuf[wid][lane][0];
  #pragma unroll
  for (int j = 0; j < 32; j += 2) {
    float a0 = fin0 * w1[j]     + fin1 * w1[32 + j]     + fin2 * w1[64 + j]     + fin3 * w1[96 + j];
    float a1 = fin0 * w1[j + 1] + fin1 * w1[32 + j + 1] + fin2 * w1[64 + j + 1] + fin3 * w1[96 + j + 1];
    hrow[j >> 1] = pack_bf16x2(fmaxf(a0, 0.0f), fmaxf(a1, 0.0f));
  }
  __syncthreads();
  int half = lane >> 4, m = lane & 15;
  #pragma unroll
  for (int ts = 0; ts < 2; ++ts) {
    v8u a32;
    const __bf16* src = &hbuf[wid][ts * 16 + m][0];
    #pragma unroll
    for (int j = 0; j < 8; ++j)
      a32[j] = *(const unsigned int*)(src + frag_kk(j, half));
    v16bf a = __builtin_bit_cast(v16bf, a32);
    #pragma unroll
    for (int nt = 0; nt < 4; ++nt) {
      const v8u* bp = (const v8u*)(p2 + ((size_t)(nt * 32 + lane) * 16));
      v16bf b = __builtin_bit_cast(v16bf, *bp);
      v8f c = {};
      c = __builtin_amdgcn_wmma_f32_16x16x32_bf16(false, a, false, b, (short)0, c, false, false);
      int n = nt * 16 + m;
      int prow = base + ts * 16;
      #pragma unroll
      for (int j = 0; j < 8; ++j) {
        int mm = j + half * 8;
        out[(size_t)(prow + mm) * 64 + n] = fmaxf(c[j], 0.0f);
      }
    }
  }
  __syncthreads();
}

__global__ void __launch_bounds__(128) point_mlp_kernel(
    const float* __restrict__ pts,
    const float* __restrict__ w1a, const __bf16* __restrict__ p2a,
    const float* __restrict__ w1b, const __bf16* __restrict__ p2b,
    const float* __restrict__ w1c, const __bf16* __restrict__ p2c,
    float* __restrict__ bone, float* __restrict__ sem, float* __restrict__ det,
    int* __restrict__ flat) {
  __shared__ __bf16 hbuf[4][32][32];
  int wid = threadIdx.x >> 5, lane = threadIdx.x & 31;
  int base = blockIdx.x * 128 + wid * 32;
  int pt = base + lane;
  __builtin_prefetch(&pts[(size_t)(pt + 128) * 5], 0, 1);
  float f0 = pts[(size_t)pt * 5 + 0];
  float f1 = pts[(size_t)pt * 5 + 1];
  float f2 = pts[(size_t)pt * 5 + 2];
  float f3 = pts[(size_t)pt * 5 + 3];
  float f4 = pts[(size_t)pt * 5 + 4];
  int bb = (int)f0;
  int xs = (int)floorf((f1 - (-51.2f)) / 0.2f);
  int ys = (int)floorf((f2 - (-51.2f)) / 0.2f);
  bool keep = (xs >= 0) && (xs < W_) && (ys >= 0) && (ys < H_);
  flat[pt] = keep ? (bb * (H_ * W_) + ys * W_ + xs) : -1;

  mlp_stage(w1a, p2a, bone, f1, f2, f3, f4, wid, lane, base, hbuf);
  mlp_stage(w1b, p2b, sem,  f1, f2, f3, f4, wid, lane, base, hbuf);
  mlp_stage(w1c, p2c, det,  f1, f2, f3, f4, wid, lane, base, hbuf);
}

// ---------------------------------------------------------------------------
// Scatter-mean accumulate (L2-resident f32 atomics)
// ---------------------------------------------------------------------------
__global__ void scatter_kernel(const float* __restrict__ bone, const int* __restrict__ flat,
                               float* __restrict__ gsum, float* __restrict__ gcnt) {
  int pt = blockIdx.x * 256 + threadIdx.x;
  if (pt >= N_) return;
  int f = flat[pt];
  if (f < 0) return;
  const float* src = bone + (size_t)pt * 64;
  float* dst = gsum + (size_t)f * 64;
  #pragma unroll 8
  for (int c = 0; c < 64; ++c) unsafeAtomicAdd(&dst[c], src[c]);
  unsafeAtomicAdd(&gcnt[f], 1.0f);
}

__global__ void mean_kernel(const float* __restrict__ gsum, const float* __restrict__ gcnt,
                            __bf16* __restrict__ gbf) {
  size_t i = (size_t)blockIdx.x * 256 + threadIdx.x;  // over S_*64
  float cnt = gcnt[i >> 6];
  gbf[i] = (__bf16)(gsum[i] / fmaxf(cnt, 1.0f));
}

// ---------------------------------------------------------------------------
// 3x3 conv, C=64 -> 64, implicit GEMM (K = 9*64 = 576 -> 18 bf16 WMMA k-steps)
// Block = 128 threads = 4 waves; block tile = 8x8 pixels; wave = 16 pixels.
// Input halo staged to LDS with gfx1250 async global->LDS loads.
// ---------------------------------------------------------------------------
#define USE_ASYNC_LDS 1

template <bool F32OUT>
__global__ void __launch_bounds__(128) conv3x3_kernel(
    const __bf16* __restrict__ in, const __bf16* __restrict__ wpk,
    __bf16* __restrict__ out_bf, float* __restrict__ out_f32) {
  __shared__ __bf16 sIn[10 * 10 * 64];
  int blk = blockIdx.x;
  int bx = blk & 63, by = (blk >> 6) & 63, bb = blk >> 12;
  int y0 = by * 8 - 1, x0 = bx * 8 - 1;
  int tid = threadIdx.x;

  // stage 10x10x64 bf16 halo as 800 x 16B chunks
  for (int c = tid; c < 800; c += 128) {
    int pix = c >> 3, sub = c & 7;
    int ly = pix / 10, lx = pix - ly * 10;
    int gy = y0 + ly, gx = x0 + lx;
    __bf16* ldst = &sIn[pix * 64 + sub * 8];
    if ((unsigned)gy < (unsigned)H_ && (unsigned)gx < (unsigned)W_) {
      const __bf16* g = in + (((size_t)(bb * H_ + gy) * W_ + gx) * 64 + sub * 8);
#if USE_ASYNC_LDS
      unsigned loff = (unsigned)(unsigned long long)(void*)ldst;
      asm volatile("global_load_async_to_lds_b128 %0, %1, off"
                   :: "v"(loff), "v"(g) : "memory");
#else
      *(uint4*)ldst = *(const uint4*)g;
#endif
    } else {
      uint4 z = {0u, 0u, 0u, 0u};
      *(uint4*)ldst = z;
    }
  }
#if USE_ASYNC_LDS
  asm volatile("s_wait_asynccnt 0" ::: "memory");
#endif
  __syncthreads();

  int wid = tid >> 5, lane = tid & 31;
  int half = lane >> 4, m = lane & 15;
  int p = wid * 16 + m;           // A-tile row -> pixel
  int pr = p >> 3, pc = p & 7;

  v8f acc[4] = {v8f{}, v8f{}, v8f{}, v8f{}};
  #pragma unroll
  for (int tap = 0; tap < 9; ++tap) {
    int dr = tap / 3, dc = tap - dr * 3;
    const __bf16* srow = &sIn[((pr + dr) * 10 + (pc + dc)) * 64];
    #pragma unroll
    for (int kc = 0; kc < 2; ++kc) {
      v8u a32;
      const __bf16* src = srow + kc * 32;
      #pragma unroll
      for (int j = 0; j < 8; ++j)
        a32[j] = *(const unsigned int*)(src + frag_kk(j, half));
      v16bf a = __builtin_bit_cast(v16bf, a32);
      int frag = tap * 2 + kc;
      #pragma unroll
      for (int nt = 0; nt < 4; ++nt) {
        const v8u* bp = (const v8u*)(wpk + ((size_t)((frag * 4 + nt) * 32 + lane) * 16));
        v16bf b = __builtin_bit_cast(v16bf, *bp);
        acc[nt] = __builtin_amdgcn_wmma_f32_16x16x32_bf16(false, a, false, b, (short)0,
                                                          acc[nt], false, false);
      }
    }
  }
  // store with ReLU; C layout: VGPR j -> M = j + half*8, N = lane&15
  #pragma unroll
  for (int nt = 0; nt < 4; ++nt) {
    #pragma unroll
    for (int j = 0; j < 8; ++j) {
      int mm = j + half * 8;
      int q = wid * 16 + mm;
      int qr = q >> 3, qc = q & 7;
      size_t off = (((size_t)(bb * H_ + by * 8 + qr) * W_ + (bx * 8 + qc)) * 64 + nt * 16 + m);
      float v = fmaxf(acc[nt][j], 0.0f);
      if (F32OUT) out_f32[off] = v;
      else        out_bf[off] = (__bf16)v;
    }
  }
}

// ---------------------------------------------------------------------------
// Classifier head: concat(cmplt[64], sem[64]) -> 64 -> 64 -> 20 (WMMA bf16)
// Block = 128 threads = 4 waves; wave = 16 points.
// ---------------------------------------------------------------------------
__global__ void __launch_bounds__(128) cls_head_kernel(
    const float* __restrict__ out_bev, const float* __restrict__ sem,
    const int* __restrict__ flat,
    const __bf16* __restrict__ pk1, const __bf16* __restrict__ pk2,
    const __bf16* __restrict__ pk3, float* __restrict__ sem_pred) {
  __shared__ __bf16 xin[4][16][128];
  __shared__ __bf16 h1[4][16][64];
  __shared__ __bf16 h2[4][16][64];
  int wid = threadIdx.x >> 5, lane = threadIdx.x & 31;
  int base = blockIdx.x * 64 + wid * 16;

  for (int idx = lane; idx < 16 * 128; idx += 32) {
    int pl = idx >> 7, ch = idx & 127;
    int pt = base + pl;
    float v;
    if (ch < 64) {
      int f = flat[pt];
      v = (f >= 0) ? out_bev[(size_t)f * 64 + ch] : 0.0f;
    } else {
      v = sem[(size_t)pt * 64 + (ch - 64)];
    }
    xin[wid][pl][ch] = (__bf16)v;
  }
  __syncthreads();
  int half = lane >> 4, m = lane & 15;

  { // layer1: K=128 -> 64
    v8f acc[4] = {v8f{}, v8f{}, v8f{}, v8f{}};
    #pragma unroll
    for (int kc = 0; kc < 4; ++kc) {
      v8u a32;
      const __bf16* src = &xin[wid][m][kc * 32];
      #pragma unroll
      for (int j = 0; j < 8; ++j)
        a32[j] = *(const unsigned int*)(src + frag_kk(j, half));
      v16bf a = __builtin_bit_cast(v16bf, a32);
      #pragma unroll
      for (int nt = 0; nt < 4; ++nt) {
        const v8u* bp = (const v8u*)(pk1 + ((size_t)((kc * 4 + nt) * 32 + lane) * 16));
        v16bf b = __builtin_bit_cast(v16bf, *bp);
        acc[nt] = __builtin_amdgcn_wmma_f32_16x16x32_bf16(false, a, false, b, (short)0,
                                                          acc[nt], false, false);
      }
    }
    #pragma unroll
    for (int nt = 0; nt < 2; ++nt) {
      unsigned* hrow;
      #pragma unroll
      for (int j = 0; j < 8; ++j) {
        // pack adjacent N columns? columns differ per lane; store as b16 pairs per row
        (void)hrow;
      }
    }
    #pragma unroll
    for (int nt = 0; nt < 4; ++nt)
      #pragma unroll
      for (int j = 0; j < 8; ++j)
        h1[wid][j + half * 8][nt * 16 + m] = (__bf16)fmaxf(acc[nt][j], 0.0f);
  }
  __syncthreads();
  { // layer2: K=64 -> 64
    v8f acc[4] = {v8f{}, v8f{}, v8f{}, v8f{}};
    #pragma unroll
    for (int kc = 0; kc < 2; ++kc) {
      v8u a32;
      const __bf16* src = &h1[wid][m][kc * 32];
      #pragma unroll
      for (int j = 0; j < 8; ++j)
        a32[j] = *(const unsigned int*)(src + frag_kk(j, half));
      v16bf a = __builtin_bit_cast(v16bf, a32);
      #pragma unroll
      for (int nt = 0; nt < 4; ++nt) {
        const v8u* bp = (const v8u*)(pk2 + ((size_t)((kc * 4 + nt) * 32 + lane) * 16));
        v16bf b = __builtin_bit_cast(v16bf, *bp);
        acc[nt] = __builtin_amdgcn_wmma_f32_16x16x32_bf16(false, a, false, b, (short)0,
                                                          acc[nt], false, false);
      }
    }
    #pragma unroll
    for (int nt = 0; nt < 4; ++nt)
      #pragma unroll
      for (int j = 0; j < 8; ++j)
        h2[wid][j + half * 8][nt * 16 + m] = (__bf16)fmaxf(acc[nt][j], 0.0f);
  }
  __syncthreads();
  { // layer3: K=64 -> 20 (padded to 32), no ReLU
    v8f acc[2] = {v8f{}, v8f{}};
    #pragma unroll
    for (int kc = 0; kc < 2; ++kc) {
      v8u a32;
      const __bf16* src = &h2[wid][m][kc * 32];
      #pragma unroll
      for (int j = 0; j < 8; ++j)
        a32[j] = *(const unsigned int*)(src + frag_kk(j, half));
      v16bf a = __builtin_bit_cast(v16bf, a32);
      #pragma unroll
      for (int nt = 0; nt < 2; ++nt) {
        const v8u* bp = (const v8u*)(pk3 + ((size_t)((kc * 2 + nt) * 32 + lane) * 16));
        v16bf b = __builtin_bit_cast(v16bf, *bp);
        acc[nt] = __builtin_amdgcn_wmma_f32_16x16x32_bf16(false, a, false, b, (short)0,
                                                          acc[nt], false, false);
      }
    }
    #pragma unroll
    for (int nt = 0; nt < 2; ++nt) {
      int n = nt * 16 + m;
      if (n < NCLS_) {
        #pragma unroll
        for (int j = 0; j < 8; ++j) {
          int pt = base + j + half * 8;
          sem_pred[(size_t)pt * NCLS_ + n] = acc[nt][j];
        }
      }
    }
  }
}

// ---------------------------------------------------------------------------
// Selection key per point (argmax / fg / sigmoid score / bg tie-break key)
// ---------------------------------------------------------------------------
__global__ void selkey_kernel(const float* __restrict__ sem_pred, float* __restrict__ key) {
  int g = blockIdx.x * 256 + threadIdx.x;
  if (g >= N_) return;
  const float* s = sem_pred + (size_t)g * NCLS_;
  float best = s[0];
  int arg = 0;
  float mx1 = -3.4e38f;
  #pragma unroll
  for (int c = 1; c < NCLS_; ++c) {
    float v = s[c];
    if (v > best) { best = v; arg = c; }
    mx1 = fmaxf(mx1, v);
  }
  bool fg = (arg > 0) && (arg < 11);
  float score = 1.0f / (1.0f + __expf(-mx1));
  int i = g % NP_;
  float bg = (float)(i % 997) * 1e-4f;
  key[g] = fg ? (1.0f + score) : bg;
}

// ---------------------------------------------------------------------------
// Deterministic exact top-k per batch: 4-round radix select on float bits
// (keys >= 0 so uint bit order == float order), then ordered compaction.
// One 1024-thread block per batch.
// ---------------------------------------------------------------------------
__global__ void __launch_bounds__(1024) topk_kernel(const float* __restrict__ key,
                                                    int* __restrict__ selidx) {
  __shared__ int hist[256];
  __shared__ unsigned sPrefix;
  __shared__ int sNeed;
  __shared__ int warpSums[32];
  __shared__ int sBase;
  int b = blockIdx.x;
  const float* k = key + (size_t)b * NP_;
  int tid = threadIdx.x;
  unsigned prefix = 0;
  int need = NPOINT_;

  for (int r = 0; r < 4; ++r) {
    int shift = 24 - 8 * r;
    if (tid < 256) hist[tid] = 0;
    __syncthreads();
    for (int i = tid; i < NP_; i += 1024) {
      unsigned bits = __float_as_uint(k[i]);
      bool ok = (r == 0) || ((bits >> (shift + 8)) == prefix);
      if (ok) atomicAdd(&hist[(bits >> shift) & 255], 1);
    }
    __syncthreads();
    if (tid == 0) {
      int cum = 0, bin = 0;
      for (int bn = 255; bn >= 0; --bn) {
        int c = hist[bn];
        if (cum + c >= need) { bin = bn; break; }
        cum += c;
      }
      sPrefix = (prefix << 8) | (unsigned)bin;
      sNeed = need - cum;
    }
    __syncthreads();
    prefix = sPrefix;
    need = sNeed;
    __syncthreads();
  }

  unsigned T = prefix;             // bits of the k-th largest key
  int ties_take = need;            // how many == T to keep (by ascending index)
  int count_gt = NPOINT_ - ties_take;
  int lane = tid & 31, wid = tid >> 5;
  unsigned lmask = (1u << lane) - 1u;
  int* outIdx = selidx + (size_t)b * NPOINT_;

  for (int phase = 0; phase < 2; ++phase) {
    if (tid == 0) sBase = (phase == 0) ? 0 : count_gt;
    __syncthreads();
    for (int i0 = 0; i0 < NP_; i0 += 1024) {
      int i = i0 + tid;
      bool pred = false;
      if (i < NP_) {
        unsigned bits = __float_as_uint(k[i]);
        pred = (phase == 0) ? (bits > T) : (bits == T);
      }
      unsigned bal = (unsigned)__ballot(pred);
      int wpre = __popc(bal & lmask);
      if (lane == 0) warpSums[wid] = __popc(bal);
      __syncthreads();
      if (tid == 0) {
        int s = sBase;
        for (int w2 = 0; w2 < 32; ++w2) { int t = warpSums[w2]; warpSums[w2] = s; s += t; }
        sBase = s;
      }
      __syncthreads();
      if (pred) {
        int pos = warpSums[wid] + wpre;
        if (pos < NPOINT_) outIdx[pos] = i;
      }
      __syncthreads();
    }
    __syncthreads();
  }
}

// ---------------------------------------------------------------------------
// Gather selected points and det features into d_out
// ---------------------------------------------------------------------------
__global__ void gather_kernel(const float* __restrict__ pts, const float* __restrict__ det,
                              const int* __restrict__ selidx,
                              float* __restrict__ out_pts, float* __restrict__ out_feat) {
  int s = blockIdx.x * 256 + threadIdx.x;
  if (s >= B_ * NPOINT_) return;
  int b = s / NPOINT_;
  int idx = selidx[s];
  size_t pt = (size_t)b * NP_ + idx;
  #pragma unroll
  for (int j = 0; j < 5; ++j) out_pts[(size_t)s * 5 + j] = pts[pt * 5 + j];
  const float* src = det + pt * 64;
  float* dst = out_feat + (size_t)s * 64;
  #pragma unroll 8
  for (int j = 0; j < 64; ++j) dst[j] = src[j];
}

// ---------------------------------------------------------------------------
extern "C" void kernel_launch(void* const* d_in, const int* in_sizes, int n_in,
                              void* d_out, int out_size, void* d_ws, size_t ws_size,
                              hipStream_t stream) {
  (void)in_sizes; (void)n_in; (void)out_size; (void)ws_size;
  const float* pts     = (const float*)d_in[0];
  const float* w_mlp1  = (const float*)d_in[1];
  const float* w_mlp2  = (const float*)d_in[2];
  const float* w_sem1  = (const float*)d_in[3];
  const float* w_sem2  = (const float*)d_in[4];
  const float* w_det1  = (const float*)d_in[5];
  const float* w_det2  = (const float*)d_in[6];
  const float* conv_w1 = (const float*)d_in[7];
  const float* conv_w2 = (const float*)d_in[8];
  const float* w_cls1  = (const float*)d_in[9];
  const float* w_cls2  = (const float*)d_in[10];
  const float* w_cls3  = (const float*)d_in[11];

  char* ws = (char*)d_ws;
  size_t o = 0;
  auto alloc = [&](size_t bytes) { size_t r = o; o += (bytes + 255) & ~(size_t)255; return r; };

  float*  det     = (float*)(ws + alloc((size_t)N_ * 64 * 4));
  float*  sem     = (float*)(ws + alloc((size_t)N_ * 64 * 4));
  float*  bone    = (float*)(ws + alloc((size_t)N_ * 64 * 4));
  int*    flat    = (int*)  (ws + alloc((size_t)N_ * 4));
  float*  gsum    = (float*)(ws + alloc((size_t)S_ * 64 * 4));  // reused as out_bev
  float*  out_bev = gsum;
  float*  gcnt    = (float*)(ws + alloc((size_t)S_ * 4));
  __bf16* gbf     = (__bf16*)(ws + alloc((size_t)S_ * 64 * 2));
  __bf16* hbf     = (__bf16*)(ws + alloc((size_t)S_ * 64 * 2));
  __bf16* pk_bone = (__bf16*)(ws + alloc(4096));
  __bf16* pk_sem  = (__bf16*)(ws + alloc(4096));
  __bf16* pk_det  = (__bf16*)(ws + alloc(4096));
  __bf16* pk_c1   = (__bf16*)(ws + alloc(73728));
  __bf16* pk_c2   = (__bf16*)(ws + alloc(73728));
  __bf16* pk_l1   = (__bf16*)(ws + alloc(16384));
  __bf16* pk_l2   = (__bf16*)(ws + alloc(8192));
  __bf16* pk_l3   = (__bf16*)(ws + alloc(4096));
  float*  key     = (float*)(ws + alloc((size_t)N_ * 4));
  int*    selidx  = (int*)  (ws + alloc((size_t)B_ * NPOINT_ * 4));

  float* out_pts  = (float*)d_out;
  float* out_feat = out_pts + (size_t)B_ * NPOINT_ * 5;
  float* sem_pred = out_feat + (size_t)B_ * NPOINT_ * 64;

  // weight packing into B-fragment layout
  pack_b_kernel<<<8,   256, 0, stream>>>(w_mlp2, pk_bone, 32, 64, 4);
  pack_b_kernel<<<8,   256, 0, stream>>>(w_sem2, pk_sem, 32, 64, 4);
  pack_b_kernel<<<8,   256, 0, stream>>>(w_det2, pk_det, 32, 64, 4);
  pack_b_kernel<<<144, 256, 0, stream>>>(conv_w1, pk_c1, 576, 64, 4);
  pack_b_kernel<<<144, 256, 0, stream>>>(conv_w2, pk_c2, 576, 64, 4);
  pack_b_kernel<<<32,  256, 0, stream>>>(w_cls1, pk_l1, 128, 64, 4);
  pack_b_kernel<<<16,  256, 0, stream>>>(w_cls2, pk_l2, 64, 64, 4);
  pack_b_kernel<<<8,   256, 0, stream>>>(w_cls3, pk_l3, 64, 20, 2);

  point_mlp_kernel<<<N_ / 128, 128, 0, stream>>>(pts, w_mlp1, pk_bone, w_sem1, pk_sem,
                                                 w_det1, pk_det, bone, sem, det, flat);

  hipMemsetAsync(gsum, 0, (size_t)S_ * 64 * 4, stream);
  hipMemsetAsync(gcnt, 0, (size_t)S_ * 4, stream);
  scatter_kernel<<<(N_ + 255) / 256, 256, 0, stream>>>(bone, flat, gsum, gcnt);
  mean_kernel<<<(S_ * 64) / 256, 256, 0, stream>>>(gsum, gcnt, gbf);

  conv3x3_kernel<false><<<B_ * 64 * 64, 128, 0, stream>>>(gbf, pk_c1, hbf, nullptr);
  conv3x3_kernel<true><<<B_ * 64 * 64, 128, 0, stream>>>(hbf, pk_c2, nullptr, out_bev);

  cls_head_kernel<<<N_ / 64, 128, 0, stream>>>(out_bev, sem, flat, pk_l1, pk_l2, pk_l3,
                                               sem_pred);
  selkey_kernel<<<(N_ + 255) / 256, 256, 0, stream>>>(sem_pred, key);
  topk_kernel<<<B_, 1024, 0, stream>>>(key, selidx);
  gather_kernel<<<(B_ * NPOINT_ + 255) / 256, 256, 0, stream>>>(pts, det, selidx,
                                                                out_pts, out_feat);
}
